// Attention_17978733101278
// MI455X (gfx1250) — matrix operand order repaired and checked by reading, not compiled, not run on hardware
//
#include <hip/hip_runtime.h>

// Problem constants (match reference: B=4, S=1024, HQ=32, HK=8, D=128)
constexpr int Bn  = 4;
constexpr int Sl  = 1024;
constexpr int HQn = 32;
constexpr int HKn = 8;
constexpr int Dh  = 128;
constexpr int G   = HQn / HKn;   // GQA group = 4
// 1/sqrt(D) * log2(e): fold softmax scale AND exp->exp2 conversion into Q
constexpr float QSCALE = 0.08838834764831845f * 1.4426950408889634f;

typedef __attribute__((ext_vector_type(16))) __bf16 v16bf;
typedef __attribute__((ext_vector_type(8)))  float  v8f;

union FragA { v16bf v; unsigned int u[8]; };

// Hardware bf16 convert (RNE)
__device__ __forceinline__ unsigned short f2bf(float a) {
  return __builtin_bit_cast(unsigned short, (__bf16)a);
}
__device__ __forceinline__ unsigned int pack2bf(float a, float b) {
  return (unsigned int)f2bf(a) | ((unsigned int)f2bf(b) << 16);
}
__device__ __forceinline__ float exp2fast(float x) {
  return __builtin_amdgcn_exp2f(x);    // v_exp_f32
}

// Max-reduce across each 16-lane row via DPP ROW_XMASK (VALU only, no LDS).
__device__ __forceinline__ float rowmax16(float x) {
#define XMASK_STEP(CTRL)                                                      \
  x = fmaxf(x, __builtin_bit_cast(float, __builtin_amdgcn_update_dpp(         \
                   0, __builtin_bit_cast(int, x), (CTRL), 0xf, 0xf, true)));
  XMASK_STEP(0x161)  // row_xmask: 1
  XMASK_STEP(0x162)  // row_xmask: 2
  XMASK_STEP(0x164)  // row_xmask: 4
  XMASK_STEP(0x168)  // row_xmask: 8
#undef XMASK_STEP
  return x;
}

__global__ __launch_bounds__(128)
void fa_fwd_causal_gqa(const float* __restrict__ q,
                       const float* __restrict__ k,
                       const float* __restrict__ v,
                       float* __restrict__ out) {
  // Double-buffered tiles: K 32 keys x 128 d (bf16 row-major),
  // V transposed [d][key].  36 KB total — tiny vs 320 KB/WGP.
  __shared__ unsigned short Ksh[2][32 * Dh];
  __shared__ unsigned short Vsh[2][Dh * 32];
  __shared__ unsigned short Psh[4][16 * 32];   // per-wave P staging (C->A)

  const int tid  = threadIdx.x;
  const int wave = tid >> 5;
  const int lane = tid & 31;
  const int ln   = lane & 15;
  const int hi   = lane >> 4;

  const int b  = blockIdx.z;
  const int hk = blockIdx.y;
  const int qb = blockIdx.x;
  const int hq = hk * G + wave;    // each wave: one query head of the GQA group
  const int q0 = qb * 16;

  // ---- Q tile as 4 bf16 A-fragments, pre-scaled by 1/sqrt(D)*log2(e) ----
  FragA aq[4];
  {
    const float* qp = q + ((size_t)(b * Sl + q0 + ln) * HQn + hq) * Dh;
#pragma unroll
    for (int c = 0; c < 4; ++c) {
#pragma unroll
      for (int vv = 0; vv < 8; ++vv) {
        const int d0 = 32 * c + 2 * (vv & 3) + ((vv & 4) << 2) + hi * 8;
        aq[c].u[vv] = pack2bf(qp[d0] * QSCALE, qp[d0 + 1] * QSCALE);
      }
    }
  }

  // All-ones B fragment: P @ ones gives per-row sum(p) replicated across lanes
  FragA ones;
#pragma unroll
  for (int vv = 0; vv < 8; ++vv) ones.u[vv] = 0x3f803f80u;   // bf16 1.0 x2

  float m_i[8];
  v8f  l_acc = {};          // row-sum accumulator (via ones-column WMMA)
  v8f  o_acc[8] = {};       // 16 rows x 128 d fp32
#pragma unroll
  for (int r = 0; r < 8; ++r) m_i[r] = -1e30f;

  const int nfull = q0 / 32;          // fully-unmasked 32-key blocks
  const int nblk  = nfull + 1;        // + diagonal block

  // ---- stage one 32-key K/V block into LDS buffer `buf` as bf16 ----
  auto stage = [&](int kb, int buf) {
    {  // K row-major; each thread: 32 contiguous d of one key -> 16 dwords
      const int key = tid >> 2, dbase = (tid & 3) * 32;
      const float* kp = k + ((size_t)(b * Sl + kb + key) * HKn + hk) * Dh + dbase;
      unsigned int* kd = (unsigned int*)&Ksh[buf][key * Dh + dbase];
#pragma unroll
      for (int i = 0; i < 8; ++i) {
        const float4 f = ((const float4*)kp)[i];
        kd[2 * i]     = pack2bf(f.x, f.y);
        kd[2 * i + 1] = pack2bf(f.z, f.w);
      }
      if (kb < nfull * 32)   // global_prefetch_b8: warms block kb+32 (= j+2)
        __builtin_prefetch(kp + 32 * HKn * Dh, 0, 1);
    }
    {  // V transposed [d][key]; each thread: a key PAIR so stores are b32
      const int k2 = (tid & 15) * 2, dbase = (tid >> 4) * 16;
      const float* va = v + ((size_t)(b * Sl + kb + k2) * HKn + hk) * Dh + dbase;
      const float* vb = va + HKn * Dh;
#pragma unroll
      for (int i = 0; i < 16; ++i)
        *(unsigned int*)&Vsh[buf][(dbase + i) * 32 + k2] = pack2bf(va[i], vb[i]);
    }
  };

  // ---- S tile = Q @ K^T for 16 keys (tile t in {0,1}) ----
  auto qk_tile = [&](int t, int buf) -> v8f {
    v8f s = {};
#pragma unroll
    for (int c = 0; c < 4; ++c) {
      FragA bk;
      const unsigned short* kt = &Ksh[buf][(t * 16 + ln) * Dh + 32 * c];
#pragma unroll
      for (int vv = 0; vv < 8; ++vv)
        bk.u[vv] = *(const unsigned int*)(kt + 2 * (vv & 3) + ((vv & 4) << 2) + hi * 8);
      s = __builtin_amdgcn_wmma_f32_16x16x32_bf16(false, aq[c].v, false, bk.v,
                                                  (short)0, s, false, false);
    }
    return s;
  };

  // ---- online softmax + P store (scores already in log2 domain) ----
  auto softmax_store = [&](v8f s0, v8f s1, bool has1, bool mask0, bool mask1,
                           int kb) {
    unsigned short* pw = Psh[wave];
#pragma unroll
    for (int r = 0; r < 8; ++r) {
      const int rowg = q0 + r + hi * 8;
      float x0 = s0[r];
      float x1 = has1 ? s1[r] : -1e30f;
      if (mask0 && (kb + ln > rowg))              x0 = -1e30f;
      if (has1 && mask1 && (kb + 16 + ln > rowg)) x1 = -1e30f;
      const float mx   = rowmax16(fmaxf(x0, x1));   // DPP row_xmask butterfly
      const float mnew = fmaxf(m_i[r], mx);
      const float cf   = exp2fast(m_i[r] - mnew);
      m_i[r] = mnew;
      const float p0 = exp2fast(x0 - mnew);
      const float p1 = has1 ? exp2fast(x1 - mnew) : 0.0f;
      l_acc[r] *= cf;
#pragma unroll
      for (int dc = 0; dc < 8; ++dc) o_acc[dc][r] *= cf;
      pw[(r + hi * 8) * 32 + ln]      = f2bf(p0);
      pw[(r + hi * 8) * 32 + ln + 16] = f2bf(p1);
    }
  };

  // ---- O += P @ V (plus ones-column WMMA for the row-sum) ----
  auto pv_accum = [&](int buf) {
    unsigned short* pw = Psh[wave];
    asm volatile("s_wait_dscnt 0" ::: "memory");   // per-wave LDS RAW fence
    FragA ap;
#pragma unroll
    for (int vv = 0; vv < 8; ++vv) {
      const int k0 = 2 * (vv & 3) + ((vv & 4) << 2) + hi * 8;
      ap.u[vv] = *(const unsigned int*)(pw + ln * 32 + k0);
    }
    l_acc = __builtin_amdgcn_wmma_f32_16x16x32_bf16(false, ap.v, false, ones.v,
                                                    (short)0, l_acc, false, false);
#pragma unroll
    for (int dc = 0; dc < 8; ++dc) {
      FragA bv;
      const unsigned short* vt = &Vsh[buf][(16 * dc + ln) * 32];
#pragma unroll
      for (int vv = 0; vv < 8; ++vv)
        bv.u[vv] = *(const unsigned int*)(vt + 2 * (vv & 3) + ((vv & 4) << 2) + hi * 8);
      o_acc[dc] = __builtin_amdgcn_wmma_f32_16x16x32_bf16(false, ap.v, false, bv.v,
                                                          (short)0, o_acc[dc],
                                                          false, false);
    }
  };

  // ---- double-buffered KV loop: one barrier per block, staging of block
  //      j+1 overlaps compute of block j ----
  stage(0, 0);
  for (int j = 0; j < nblk; ++j) {
    const int buf = j & 1;
    const int kb  = 32 * j;
    __syncthreads();                       // buf staged; buf^1 fully consumed
    if (j + 1 < nblk) stage(kb + 32, buf ^ 1);

    v8f s0 = qk_tile(0, buf);
    if (j < nfull) {                       // fully-unmasked block
      v8f s1 = qk_tile(1, buf);
      softmax_store(s0, s1, true, false, false, kb);
    } else if (q0 & 16) {                  // diag: tile0 full, tile1 triangular
      v8f s1 = qk_tile(1, buf);
      softmax_store(s0, s1, true, false, true, kb);
    } else {                               // diag: tile0 triangular, tile1 all-masked
      v8f z = {};
      softmax_store(s0, z, false, true, false, kb);
    }
    pv_accum(buf);
  }

  // ---- epilogue: out = O / l ----
  float inv_l[8];
#pragma unroll
  for (int r = 0; r < 8; ++r) inv_l[r] = 1.0f / l_acc[r];
#pragma unroll
  for (int dc = 0; dc < 8; ++dc) {
#pragma unroll
    for (int r = 0; r < 8; ++r) {
      const int row = q0 + r + hi * 8;
      const int d   = 16 * dc + ln;
      out[((size_t)(b * Sl + row) * HQn + hq) * Dh + d] = o_acc[dc][r] * inv_l[r];
    }
  }
}

extern "C" void kernel_launch(void* const* d_in, const int* in_sizes, int n_in,
                              void* d_out, int out_size, void* d_ws, size_t ws_size,
                              hipStream_t stream) {
  const float* q = (const float*)d_in[0];
  const float* k = (const float*)d_in[1];
  const float* v = (const float*)d_in[2];
  float* out = (float*)d_out;
  (void)in_sizes; (void)n_in; (void)out_size; (void)d_ws; (void)ws_size;

  dim3 grid(Sl / 16, HKn, Bn);   // (64 q-blocks, 8 kv-heads, 4 batches)
  dim3 block(128);               // 4 waves: one per GQA query head
  fa_fwd_causal_gqa<<<grid, block, 0, stream>>>(q, k, v, out);
}
